// CustomCenterQuantizerLinear_20486994002248
// MI455X (gfx1250) — compile-verified, matrix operand order checked by reading, not run on hardware
//
#include <hip/hip_runtime.h>

// ---------------------------------------------------------------------------
// Fused CenterQuantizer dequant + GEMM for MI455X (gfx1250, wave32, WMMA)
//   y[256,8192] = x[256,8192] @ f(Wq)[8192,8192]^T + bias
// Roofline: 256 MB int32 weight stream / 23.3 TB/s ~= 11 us floor.
// Weights read from HBM exactly once, dequantized in registers, staged as
// bf16 in double-buffered LDS (80 KB), consumed by v_wmma_f32_16x16x32_bf16.
// A tiles move global->LDS via GLOBAL_LOAD_ASYNC_TO_LDS_B128 (ASYNCcnt).
// 8 waves/block = 2 waves/SIMD so hardware SMT hides ds_load latency that
// the allocator would otherwise serialize against the WMMA chain.
// ---------------------------------------------------------------------------

typedef __attribute__((ext_vector_type(16))) __bf16 v16bf;
typedef __attribute__((ext_vector_type(8)))  float  v8f;
typedef __attribute__((ext_vector_type(4)))  int    v4i;

typedef __attribute__((address_space(1))) v4i* gptr4;   // global int4
typedef __attribute__((address_space(3))) v4i* lptr4;   // LDS int4

#define IN_DIM  8192
#define OUT_DIM 8192
#define M_DIM   256
#define KTILE   64
#define NTILE   64

#if __has_builtin(__builtin_amdgcn_global_load_async_to_lds_b128)
#define HAVE_ASYNC_LDS 1
#endif
#if __has_builtin(__builtin_amdgcn_s_wait_asynccnt)
#define HAVE_WAIT_ASYNC 1
#endif

union Frag {
  uint4 u4[2];
  v16bf v;
};

__device__ __forceinline__ unsigned int pack2_bf16(float lo, float hi) {
  unsigned int ul = __float_as_uint(lo);
  ul += 0x7FFFu + ((ul >> 16) & 1u);          // round-to-nearest-even
  unsigned int uh = __float_as_uint(hi);
  uh += 0x7FFFu + ((uh >> 16) & 1u);
  return (ul >> 16) | (uh & 0xFFFF0000u);
}

__device__ __forceinline__ float dequant(int q, float inv_sc, float eps, float gam) {
  float y = (float)q * inv_sc;
  float a = fabsf(y);
  float v = (a > 1.0f) ? (gam * __expf(a - 1.0f))
                       : (eps + a * (gam - eps));
  v = copysignf(v, y);
  return (q == 0) ? 0.0f : v;
}

__device__ __forceinline__ void load_w(int4 w[4], const int* wp) {
  const int4* p = (const int4*)wp;
  w[0] = p[0]; w[1] = p[1]; w[2] = p[2]; w[3] = p[3];
}

// Thread t owns weight column `wcol`, K-quarter `q` (16 ints = 32 bf16 bytes).
__device__ __forceinline__ void dequant_store(const int4 w[4], uint4* sBb,
                                              int wcol, int q,
                                              float inv_sc, float eps, float gam) {
  int wv[16] = {w[0].x, w[0].y, w[0].z, w[0].w, w[1].x, w[1].y, w[1].z, w[1].w,
                w[2].x, w[2].y, w[2].z, w[2].w, w[3].x, w[3].y, w[3].z, w[3].w};
  unsigned int pk[8];
#pragma unroll
  for (int i = 0; i < 8; ++i)
    pk[i] = pack2_bf16(dequant(wv[2 * i],     inv_sc, eps, gam),
                       dequant(wv[2 * i + 1], inv_sc, eps, gam));
  sBb[wcol * 8 + q * 2 + 0] = make_uint4(pk[0], pk[1], pk[2], pk[3]);
  sBb[wcol * 8 + q * 2 + 1] = make_uint4(pk[4], pk[5], pk[6], pk[7]);
}

// ---------------------------------------------------------------------------
// Kernel 1: x (f32) -> bf16 activation matrix in workspace. 8 floats/thread.
// ---------------------------------------------------------------------------
__global__ void x_to_bf16(const float* __restrict__ x, uint4* __restrict__ a) {
  int t = blockIdx.x * blockDim.x + threadIdx.x;
  const float4* x4 = (const float4*)x;
  float4 f0 = x4[2 * t];
  float4 f1 = x4[2 * t + 1];
  uint4 o;
  o.x = pack2_bf16(f0.x, f0.y);
  o.y = pack2_bf16(f0.z, f0.w);
  o.z = pack2_bf16(f1.x, f1.y);
  o.w = pack2_bf16(f1.z, f1.w);
  a[t] = o;
}

// ---------------------------------------------------------------------------
// Kernel 2: fused dequant + GEMM.
// Grid: 128 blocks (one per 64 output columns), 256 threads = 8 waves.
// Wave (wm, wv3): wm = M-half (128 rows = 8 WMMA tiles), wv3 = 16-col slice.
// ---------------------------------------------------------------------------
__global__ void __launch_bounds__(256, 1)
fused_dequant_gemm(const unsigned short* __restrict__ A,   // bf16 [256][8192]
                   const int*            __restrict__ Wq,  // int32 [8192][8192]
                   const float*          __restrict__ epsp,
                   const float*          __restrict__ gamp,
                   const float*          __restrict__ scp,
                   const float*          __restrict__ bias,
                   float*                __restrict__ out) {
  __shared__ uint4 sA[2][M_DIM * 8];   // 2 x 32 KB : 256 rows x 64 K bf16
  __shared__ uint4 sB[2][NTILE * 8];   // 2 x  8 KB :  64 cols x 64 K bf16

  const int t    = threadIdx.x;
  const int lane = t & 31;
  const int wave = t >> 5;
  const int lp   = lane & 15;   // M (A) / N (B) index within tile
  const int h    = lane >> 4;   // K-half select per ISA 16-bit layout
  const int wv3  = wave & 3;    // column slice 0..3
  const int wm   = wave >> 2;   // M half 0..1
  const int n0   = blockIdx.x * NTILE;

  const float eps    = epsp[0];
  const float gam    = gamp[0];
  const float inv_sc = 1.0f / scp[0];

  v8f acc[8];
  {
    v8f zero = {0.f, 0.f, 0.f, 0.f, 0.f, 0.f, 0.f, 0.f};
#pragma unroll
    for (int m = 0; m < 8; ++m) acc[m] = zero;
  }

  // weight-tile assignment: thread t -> column wcol, K-quarter q (16 ints)
  const int wcol = t >> 2;
  const int q    = t & 3;
  const int* wp0 = Wq + (size_t)(n0 + wcol) * IN_DIM + q * 16;

  int4 w[4];
#if !defined(HAVE_ASYNC_LDS)
  uint4 areg[8];
#endif

  // ---- A-tile staging: 2048 x 16B chunks, 8 per thread ----
#if defined(HAVE_ASYNC_LDS)
#define ISSUE_A_TILE(k0_, dst_)                                               \
  {                                                                           \
    _Pragma("unroll") for (int i = 0; i < 8; ++i) {                           \
      int c_ = t + i * 256, row_ = c_ >> 3, q8_ = c_ & 7;                     \
      const uint4* src_ = (const uint4*)(A + (size_t)row_ * IN_DIM + (k0_) + q8_ * 8); \
      __builtin_amdgcn_global_load_async_to_lds_b128(                         \
          (gptr4)src_, (lptr4)((dst_) + row_ * 8 + q8_), 0, 0);               \
    }                                                                         \
  }
#define FINISH_A_TILE(dst_) /* data lands directly in LDS */
#else
#define ISSUE_A_TILE(k0_, dst_)                                               \
  {                                                                           \
    _Pragma("unroll") for (int i = 0; i < 8; ++i) {                           \
      int c_ = t + i * 256, row_ = c_ >> 3, q8_ = c_ & 7;                     \
      areg[i] = *(const uint4*)(A + (size_t)row_ * IN_DIM + (k0_) + q8_ * 8); \
    }                                                                         \
  }
#define FINISH_A_TILE(dst_)                                                   \
  {                                                                           \
    _Pragma("unroll") for (int i = 0; i < 8; ++i) {                           \
      int c_ = t + i * 256, row_ = c_ >> 3, q8_ = c_ & 7;                     \
      (dst_)[row_ * 8 + q8_] = areg[i];                                       \
    }                                                                         \
  }
#endif

  // ---- prologue: stage k0 = 0 into buffer 0 ----
  load_w(w, wp0);
  ISSUE_A_TILE(0, sA[0]);
  dequant_store(w, sB[0], wcol, q, inv_sc, eps, gam);
  FINISH_A_TILE(sA[0]);

  const int bcol  = wv3 * 16 + lp;           // B-fragment column
  const int arow0 = wm * 8 * 16 + lp;        // first A-fragment row
  int buf = 0;

  for (int k0 = 0; k0 < IN_DIM; k0 += KTILE) {
#if defined(HAVE_ASYNC_LDS) && defined(HAVE_WAIT_ASYNC)
    __builtin_amdgcn_s_wait_asynccnt(0);
#endif
    __syncthreads();                       // publish tile `buf`

    const int nb    = buf ^ 1;
    const bool more = (k0 + KTILE) < IN_DIM;
    if (more) {
      load_w(w, wp0 + k0 + KTILE);         // HBM latency hidden under WMMAs
      ISSUE_A_TILE(k0 + KTILE, sA[nb]);
      if (k0 + 2 * KTILE < IN_DIM)
        __builtin_prefetch(wp0 + k0 + 2 * KTILE, 0, 1);  // global_prefetch_b8
    }

    // ---- B fragments for both K-substeps (ISA 16-bit B layout) ----
    Frag bf0, bf1;
    bf0.u4[0] = sB[buf][bcol * 8 + h];
    bf0.u4[1] = sB[buf][bcol * 8 + 2 + h];
    bf1.u4[0] = sB[buf][bcol * 8 + 4 + h];
    bf1.u4[1] = sB[buf][bcol * 8 + 6 + h];

    // ---- 2 K-substeps x 8 M-tiles; A fragments double-buffered ----
#pragma unroll
    for (int kk = 0; kk < 2; ++kk) {
      const int kb = kk * 4;
      Frag af[2];
      af[0].u4[0] = sA[buf][arow0 * 8 + kb + h];
      af[0].u4[1] = sA[buf][arow0 * 8 + kb + 2 + h];
#pragma unroll
      for (int m = 0; m < 8; ++m) {
        if (m < 7) {
          int ridx = (arow0 + (m + 1) * 16) * 8 + kb;
          af[(m + 1) & 1].u4[0] = sA[buf][ridx + h];
          af[(m + 1) & 1].u4[1] = sA[buf][ridx + 2 + h];
        }
        acc[m] = __builtin_amdgcn_wmma_f32_16x16x32_bf16(
            false, af[m & 1].v, false, kk ? bf1.v : bf0.v,
            (short)0, acc[m], false, false);
      }
    }

    if (more) {
      // dequant VALU co-executes with the other wave's XDL bursts
      dequant_store(w, sB[nb], wcol, q, inv_sc, eps, gam);
      FINISH_A_TILE(sA[nb]);
    }
    buf = nb;
  }

  // ---- epilogue: C layout is VGPR v -> row tile*16 + v + 8*h, lane -> col ----
  const int   col = n0 + wv3 * 16 + lp;
  const float bv  = bias[col];
#pragma unroll
  for (int m = 0; m < 8; ++m) {
#pragma unroll
    for (int v = 0; v < 8; ++v) {
      int row = (wm * 8 + m) * 16 + v + 8 * h;
      out[(size_t)row * OUT_DIM + col] = acc[m][v] + bv;
    }
  }
}

// ---------------------------------------------------------------------------
extern "C" void kernel_launch(void* const* d_in, const int* in_sizes, int n_in,
                              void* d_out, int out_size, void* d_ws, size_t ws_size,
                              hipStream_t stream) {
  const float* x    = (const float*)d_in[0];
  const float* eps  = (const float*)d_in[1];
  const float* gam  = (const float*)d_in[2];
  const float* sc   = (const float*)d_in[3];
  const float* bias = (const float*)d_in[4];
  const int*   wq   = (const int*)d_in[5];
  float*       out  = (float*)d_out;

  unsigned short* A_bf = (unsigned short*)d_ws;   // 256*8192 bf16 = 4 MB

  // 256*8192 floats / 8 per thread = 262144 threads
  x_to_bf16<<<1024, 256, 0, stream>>>(x, (uint4*)A_bf);

  fused_dequant_gemm<<<OUT_DIM / NTILE, 256, 0, stream>>>(
      A_bf, wq, eps, gam, sc, bias, out);
}
